// DecoderLayer_54855322304772
// MI455X (gfx1250) — matrix operand order repaired and checked by reading, not compiled, not run on hardware
//
#include <hip/hip_runtime.h>
#include <hip/hip_bf16.h>
#include <stdint.h>

// ---------------------------------------------------------------------------
// DecoderLayer for MI455X (gfx1250).
//  - All matmuls: v_wmma_f32_16x16x32_bf16 (bf16 operands, fp32 accumulate).
//  - GEMM operand staging: Tensor Data Mover (tensor_load_to_lds), double
//    buffered with s_wait_tensorcnt so DMA overlaps WMMA compute.
//  - B=4, S=4096, D=1024, H=16, DH=64, BLEN=256, FF=4096.
//  - Workspace requirement: ~344 MB.
// ---------------------------------------------------------------------------

typedef __bf16 bf16_t;
typedef __attribute__((ext_vector_type(4)))  __bf16 v4bf;
typedef __attribute__((ext_vector_type(8)))  __bf16 v8bf;
typedef __attribute__((ext_vector_type(16))) __bf16 v16bf;
typedef __attribute__((ext_vector_type(8)))  float  v8f;
typedef __attribute__((ext_vector_type(4)))  unsigned int u32x4;
typedef __attribute__((ext_vector_type(4)))  int i32x4;
typedef __attribute__((ext_vector_type(8)))  int i32x8;

union Frag { v16bf v; struct { v8bf lo, hi; } p; };

__device__ __forceinline__ v8f wmma_bf16(const v16bf& a, const v16bf& b, const v8f& c) {
    return __builtin_amdgcn_wmma_f32_16x16x32_bf16(false, a, false, b, (short)0, c, false, false);
}

__device__ __forceinline__ uint32_t lds_offset(const void* p) {
    return (uint32_t)(uintptr_t)p;   // low 32 bits of generic addr = LDS offset
}

#if __has_include(<hip/amd_detail/amd_gfx1250_TDM.h>)
#define TDM_ISSUE(g0, g1, za, zb)                                             \
    do { const i32x8 z8_ = {0,0,0,0,0,0,0,0};                                 \
         __builtin_amdgcn_tensor_load_to_lds(g0, g1, za, zb, z8_, 0); } while (0)
#else
#define TDM_ISSUE(g0, g1, za, zb)                                             \
    __builtin_amdgcn_tensor_load_to_lds(g0, g1, za, zb, 0)
#endif

// TDM: DMA a (tile_h x tile_w) bf16 tile, row stride `stride` elements, into LDS.
__device__ __forceinline__ void tdm_load_2d_bf16(uint32_t lds_addr, const bf16_t* gaddr,
                                                 uint32_t tile_w, uint32_t tile_h,
                                                 uint32_t stride) {
    const uint64_t ga = (uint64_t)(uintptr_t)gaddr;
    u32x4 g0;
    g0[0] = 1u;                                   // count=1, user descriptor
    g0[1] = lds_addr;                             // LDS byte address
    g0[2] = (uint32_t)ga;                         // global_addr[31:0]
    g0[3] = (uint32_t)(ga >> 32) | (2u << 30);    // global_addr[56:32] | type=2
    i32x8 g1;
    g1[0] = (int)(1u << 16);                      // data_size=1 (2B); wg_mask=0
    g1[1] = (int)((stride & 0xFFFFu) << 16);      // tensor_dim0[15:0]  (=stride)
    g1[2] = (int)((stride >> 16) & 0xFFFFu);      // tensor_dim0[31:16]; dim1 lo16=0
    g1[3] = (int)(16u | (tile_w << 16));          // tensor_dim1=1<<20 (hi16); tile_dim0
    g1[4] = (int)tile_h;                          // tile_dim1; tile_dim2=0
    g1[5] = (int)stride;                          // tensor_dim0_stride[31:0]
    g1[6] = 0;                                    // stride hi; dim1_stride lo
    g1[7] = 0;
    const i32x4 z4 = {0, 0, 0, 0};
    TDM_ISSUE(g0, g1, z4, z4);
}

// ---------------------------------------------------------------------------
// One-time fp32 -> bf16 flat convert (vectorized).
// ---------------------------------------------------------------------------
__global__ __launch_bounds__(256) void convert_f32_bf16(
    const float* __restrict__ in, bf16_t* __restrict__ out, long long n)
{
    const size_t i = ((size_t)blockIdx.x * 256 + threadIdx.x) * 8;
    if ((long long)(i + 8) <= n) {
        const float4 a = *(const float4*)(in + i);
        const float4 b = *(const float4*)(in + i + 4);
        v8bf p;
        p[0] = (bf16_t)a.x; p[1] = (bf16_t)a.y; p[2] = (bf16_t)a.z; p[3] = (bf16_t)a.w;
        p[4] = (bf16_t)b.x; p[5] = (bf16_t)b.y; p[6] = (bf16_t)b.z; p[7] = (bf16_t)b.w;
        *(v8bf*)(out + i) = p;
    }
}

// ---------------------------------------------------------------------------
// One-time weight transpose-convert: W[K][N] fp32 -> Wt[N][K] bf16 (64x64 tiles).
// ---------------------------------------------------------------------------
__global__ __launch_bounds__(256) void transpose_convert(
    const float* __restrict__ W, bf16_t* __restrict__ Wt, int K, int N)
{
    __shared__ float sT[64][68];                 // padded vs bank conflicts
    const int bk = blockIdx.y * 64;
    const int bn = blockIdx.x * 64;
    const int tid = threadIdx.x;
    const int r  = tid >> 4;                     // 0..15
    const int c4 = (tid & 15) * 4;               // 0..60
#pragma unroll
    for (int i = 0; i < 4; ++i) {
        const int k = r + i * 16;
        const float4 v = *(const float4*)(W + (size_t)(bk + k) * N + bn + c4);
        sT[k][c4 + 0] = v.x; sT[k][c4 + 1] = v.y; sT[k][c4 + 2] = v.z; sT[k][c4 + 3] = v.w;
    }
    __syncthreads();
#pragma unroll
    for (int i = 0; i < 4; ++i) {
        const int n = r + i * 16;
        v4bf p;
        p[0] = (bf16_t)sT[c4 + 0][n]; p[1] = (bf16_t)sT[c4 + 1][n];
        p[2] = (bf16_t)sT[c4 + 2][n]; p[3] = (bf16_t)sT[c4 + 3][n];
        *(v4bf*)&Wt[(size_t)(bn + n) * K + bk + c4] = p;
    }
}

// ---------------------------------------------------------------------------
// Tiled GEMM: C[M,N] = act(alpha * A[M,K] @ Wt[N,K]^T + bias)
// bf16 operands, TDM double-buffered staging, 8 waves, wave tile 32x64.
// ---------------------------------------------------------------------------
template <bool BF16OUT>
__global__ __launch_bounds__(256) void gemm_tdm_wmma(
    const bf16_t* __restrict__ A,   // [M][K] bf16
    const bf16_t* __restrict__ Wt,  // [N][K] bf16 (pre-transposed)
    const float*  __restrict__ bias,
    void* __restrict__ Cp,
    int M, int N, int K, float alpha, int relu)
{
    constexpr int BM = 128, BN = 128, BK = 32;
    __shared__ __align__(16) bf16_t sA[2][BM][BK];
    __shared__ __align__(16) bf16_t sB[2][BN][BK];

    const int tid  = threadIdx.x;
    const int wid  = tid >> 5;
    const int lane = tid & 31;
    const int half = lane >> 4;
    const int l16  = lane & 15;
    const int bm   = blockIdx.y * BM;
    const int bn   = blockIdx.x * BN;
    const int wm   = (wid & 3) * 32;
    const int wn   = (wid >> 2) * 64;

    const uint32_t ldsA[2] = { lds_offset(&sA[0][0][0]), lds_offset(&sA[1][0][0]) };
    const uint32_t ldsB[2] = { lds_offset(&sB[0][0][0]), lds_offset(&sB[1][0][0]) };

    v8f acc[2][4] = {};
    const int nsteps = K / BK;

    if (wid == 0) {   // prologue: DMA tiles for step 0
        tdm_load_2d_bf16(ldsA[0], A  + (size_t)bm * K, BK, BM, K);
        tdm_load_2d_bf16(ldsB[0], Wt + (size_t)bn * K, BK, BN, K);
    }

    for (int s = 0; s < nsteps; ++s) {
        const int cur = s & 1;
        __syncthreads();                          // buffer cur^1 free for DMA
        if (wid == 0) {
            if (s + 1 < nsteps) {                 // prefetch next tiles via TDM
                const size_t k1 = (size_t)(s + 1) * BK;
                tdm_load_2d_bf16(ldsA[cur ^ 1], A  + (size_t)bm * K + k1, BK, BM, K);
                tdm_load_2d_bf16(ldsB[cur ^ 1], Wt + (size_t)bn * K + k1, BK, BN, K);
                __builtin_amdgcn_s_wait_tensorcnt(2);  // current 2 DMAs done
            } else {
                __builtin_amdgcn_s_wait_tensorcnt(0);
            }
        }
        __syncthreads();                          // buffer cur visible to all

        Frag a[2];
#pragma unroll
        for (int t = 0; t < 2; ++t) {
            const int row = wm + t * 16 + l16;
            const int kb  = half * 8;
            a[t].p.lo = *(const v8bf*)&sA[cur][row][kb];
            a[t].p.hi = *(const v8bf*)&sA[cur][row][kb + 16];
        }
#pragma unroll
        for (int u = 0; u < 4; ++u) {
            Frag b;
            const int col = wn + u * 16 + l16;
            const int kb2 = half * 8;
            b.p.lo = *(const v8bf*)&sB[cur][col][kb2];
            b.p.hi = *(const v8bf*)&sB[cur][col][kb2 + 16];
#pragma unroll
            for (int t = 0; t < 2; ++t)
                acc[t][u] = wmma_bf16(a[t].v, b.v, acc[t][u]);
        }
    }

#pragma unroll
    for (int t = 0; t < 2; ++t)
#pragma unroll
        for (int u = 0; u < 4; ++u) {
            const int col  = bn + wn + u * 16 + l16;
            const float bv = bias ? bias[col] : 0.0f;
            const int row0 = bm + wm + t * 16 + half * 8;
#pragma unroll
            for (int v = 0; v < 8; ++v) {
                float x = acc[t][u][v] * alpha + bv;
                if (relu) x = fmaxf(x, 0.0f);
                if constexpr (BF16OUT)
                    ((bf16_t*)Cp)[(size_t)(row0 + v) * N + col] = (bf16_t)x;
                else
                    ((float*)Cp)[(size_t)(row0 + v) * N + col] = x;
            }
        }
}

// ---------------------------------------------------------------------------
// Local block attention (flash-style online softmax), bf16 in/out.
// Q tile + K chunks DMA'd by TDM; V staged transposed manually.
// grid = (nb=16, B*H=64), 256 threads (8 waves x 32 q rows).
// ---------------------------------------------------------------------------
__global__ __launch_bounds__(256) void local_attn_wmma(
    const bf16_t* __restrict__ Qg, const bf16_t* __restrict__ Kg,
    const bf16_t* __restrict__ Vg, bf16_t* __restrict__ Og)
{
    constexpr int Sq = 4096, Dm = 1024, DH = 64;
    const int blk = blockIdx.x;
    const int b   = blockIdx.y >> 4;
    const int h   = blockIdx.y & 15;

    __shared__ __align__(16) bf16_t sQ[256][64];
    __shared__ __align__(16) bf16_t sK[64][64];
    __shared__ __align__(16) bf16_t sVt[64][64];
    __shared__ __align__(16) bf16_t sP[8][16][32];

    const int tid  = threadIdx.x;
    const int wid  = tid >> 5;
    const int lane = tid & 31;
    const int half = lane >> 4;
    const int l16  = lane & 15;

    const int win0 = (blk == 0) ? 0 : (blk - 1) * 256;
    const int nch  = (blk == 0) ? 4 : 8;

    if (wid == 0) {   // DMA whole scaled-Q block (scale folded into Q GEMM alpha)
        tdm_load_2d_bf16(lds_offset(&sQ[0][0]),
                         Qg + ((size_t)b * Sq + blk * 256) * Dm + h * DH,
                         DH, 256, Dm);
    }

    v8f O[2][4] = {};
    float rmax[2][8], rsum[2][8];
#pragma unroll
    for (int t = 0; t < 2; ++t)
#pragma unroll
        for (int v = 0; v < 8; ++v) { rmax[t][v] = -3.0e38f; rsum[t][v] = 0.0f; }

    for (int ch = 0; ch < nch; ++ch) {
        __syncthreads();                          // prev chunk compute done
        if (wid == 0) {                           // DMA K chunk
            tdm_load_2d_bf16(lds_offset(&sK[0][0]),
                             Kg + ((size_t)b * Sq + win0 + ch * 64) * Dm + h * DH,
                             DH, 64, Dm);
        }
        {   // stage V chunk transposed: load first, then store (batched)
            const int r = tid >> 2;
            const int c = (tid & 3) * 16;
            const bf16_t* pv = Vg + ((size_t)b * Sq + win0 + ch * 64 + r) * Dm + h * DH + c;
            const v8bf a0 = *(const v8bf*)pv;
            const v8bf a1 = *(const v8bf*)(pv + 8);
#pragma unroll
            for (int j = 0; j < 8; ++j) {
                sVt[c + j][r]     = a0[j];
                sVt[c + 8 + j][r] = a1[j];
            }
        }
        if (wid == 0) __builtin_amdgcn_s_wait_tensorcnt(0);  // K (and Q) done
        __syncthreads();

#pragma unroll
        for (int js = 0; js < 2; ++js) {          // two 32-k steps per chunk
#pragma unroll
            for (int t = 0; t < 2; ++t) {         // wave's two q-tiles
                v8f s0 = {}, s1 = {};
#pragma unroll
                for (int dk = 0; dk < 64; dk += 32) {
                    Frag a, b0, b1;
                    const int qrow = wid * 32 + t * 16 + l16;
                    a.p.lo = *(const v8bf*)&sQ[qrow][dk + half * 8];
                    a.p.hi = *(const v8bf*)&sQ[qrow][dk + half * 8 + 16];
                    const int kr0 = js * 32 + l16;
                    b0.p.lo = *(const v8bf*)&sK[kr0][dk + half * 8];
                    b0.p.hi = *(const v8bf*)&sK[kr0][dk + half * 8 + 16];
                    const int kr1 = kr0 + 16;
                    b1.p.lo = *(const v8bf*)&sK[kr1][dk + half * 8];
                    b1.p.hi = *(const v8bf*)&sK[kr1][dk + half * 8 + 16];
                    s0 = wmma_bf16(a.v, b0.v, s0);
                    s1 = wmma_bf16(a.v, b1.v, s1);
                }
                const int qbase = blk * 256 + wid * 32 + t * 16 + half * 8;
                const int ka0   = win0 + ch * 64 + js * 32 + l16;
                const int ka1   = ka0 + 16;
                float tmax[8];
#pragma unroll
                for (int v = 0; v < 8; ++v) {
                    const int qa = qbase + v;
                    if (ka0 > qa) s0[v] = -1.0e9f;
                    if (ka1 > qa) s1[v] = -1.0e9f;
                    tmax[v] = fmaxf(s0[v], s1[v]);
                }
#pragma unroll
                for (int m = 8; m >= 1; m >>= 1)
#pragma unroll
                    for (int v = 0; v < 8; ++v)
                        tmax[v] = fmaxf(tmax[v], __shfl_xor(tmax[v], m));
                float corr[8], tsum[8];
#pragma unroll
                for (int v = 0; v < 8; ++v) {
                    const float nm = fmaxf(rmax[t][v], tmax[v]);
                    corr[v]    = __expf(rmax[t][v] - nm);
                    rmax[t][v] = nm;
                    s0[v] = __expf(s0[v] - nm);
                    s1[v] = __expf(s1[v] - nm);
                    tsum[v] = s0[v] + s1[v];
                }
#pragma unroll
                for (int m = 8; m >= 1; m >>= 1)
#pragma unroll
                    for (int v = 0; v < 8; ++v)
                        tsum[v] += __shfl_xor(tsum[v], m);
#pragma unroll
                for (int v = 0; v < 8; ++v)
                    rsum[t][v] = rsum[t][v] * corr[v] + tsum[v];
#pragma unroll
                for (int u = 0; u < 4; ++u)
#pragma unroll
                    for (int v = 0; v < 8; ++v) O[t][u][v] *= corr[v];

                // Re-layout P (D layout -> A layout) via per-wave LDS scratch
#pragma unroll
                for (int v = 0; v < 8; ++v) {
                    sP[wid][v + 8 * half][l16]      = (bf16_t)s0[v];
                    sP[wid][v + 8 * half][16 + l16] = (bf16_t)s1[v];
                }
                asm volatile("s_wait_dscnt 0" ::: "memory");  // intra-wave LDS RAW

                Frag pa;
                pa.p.lo = *(const v8bf*)&sP[wid][l16][half * 8];
                pa.p.hi = *(const v8bf*)&sP[wid][l16][half * 8 + 16];
#pragma unroll
                for (int u = 0; u < 4; ++u) {     // O += P @ V
                    Frag vb;
                    const int dcol = u * 16 + l16;
                    vb.p.lo = *(const v8bf*)&sVt[dcol][js * 32 + half * 8];
                    vb.p.hi = *(const v8bf*)&sVt[dcol][js * 32 + half * 8 + 16];
                    O[t][u] = wmma_bf16(pa.v, vb.v, O[t][u]);
                }
            }
        }
    }

#pragma unroll
    for (int t = 0; t < 2; ++t)
#pragma unroll
        for (int v = 0; v < 8; ++v) {
            const float inv = 1.0f / rsum[t][v];
            const int qr = blk * 256 + wid * 32 + t * 16 + half * 8 + v;
#pragma unroll
            for (int u = 0; u < 4; ++u) {
                const int d = u * 16 + l16;
                Og[((size_t)b * Sq + qr) * Dm + h * DH + d] = (bf16_t)(O[t][u][v] * inv);
            }
        }
}

// ---------------------------------------------------------------------------
// out = LayerNorm(Y + X) * g + b ; writes fp32 and (optionally) bf16 copies.
// ---------------------------------------------------------------------------
__global__ __launch_bounds__(256) void add_layernorm(
    const float* __restrict__ Y, const float* __restrict__ X,
    const float* __restrict__ g, const float* __restrict__ bb,
    float* __restrict__ outf, bf16_t* __restrict__ outb)
{
    const int row = blockIdx.x;
    const int tid = threadIdx.x;
    const float4 a = *(const float4*)(Y + (size_t)row * 1024 + tid * 4);
    const float4 c = *(const float4*)(X + (size_t)row * 1024 + tid * 4);
    const float v0 = a.x + c.x, v1 = a.y + c.y, v2 = a.z + c.z, v3 = a.w + c.w;
    float s = v0 + v1 + v2 + v3;
    float q = v0 * v0 + v1 * v1 + v2 * v2 + v3 * v3;
#pragma unroll
    for (int m = 16; m >= 1; m >>= 1) { s += __shfl_xor(s, m); q += __shfl_xor(q, m); }
    __shared__ float ssum[8], ssq[8];
    const int wid = tid >> 5;
    if ((tid & 31) == 0) { ssum[wid] = s; ssq[wid] = q; }
    __syncthreads();
    float ts = 0.0f, tq = 0.0f;
#pragma unroll
    for (int i = 0; i < 8; ++i) { ts += ssum[i]; tq += ssq[i]; }
    const float mu  = ts * (1.0f / 1024.0f);
    const float var = tq * (1.0f / 1024.0f) - mu * mu;
    const float r   = rsqrtf(var + 1e-6f);
    const int c0 = tid * 4;
    const float o0 = (v0 - mu) * r * g[c0 + 0] + bb[c0 + 0];
    const float o1 = (v1 - mu) * r * g[c0 + 1] + bb[c0 + 1];
    const float o2 = (v2 - mu) * r * g[c0 + 2] + bb[c0 + 2];
    const float o3 = (v3 - mu) * r * g[c0 + 3] + bb[c0 + 3];
    float* po = outf + (size_t)row * 1024 + c0;
    po[0] = o0; po[1] = o1; po[2] = o2; po[3] = o3;
    if (outb) {
        v4bf p;
        p[0] = (bf16_t)o0; p[1] = (bf16_t)o1; p[2] = (bf16_t)o2; p[3] = (bf16_t)o3;
        *(v4bf*)(outb + (size_t)row * 1024 + c0) = p;
    }
}

// ---------------------------------------------------------------------------
extern "C" void kernel_launch(void* const* d_in, const int* in_sizes, int n_in,
                              void* d_out, int out_size, void* d_ws, size_t ws_size,
                              hipStream_t stream) {
    const float* X      = (const float*)d_in[0];
    const float* Wq     = (const float*)d_in[1];
    const float* Wk     = (const float*)d_in[2];
    const float* Wv     = (const float*)d_in[3];
    const float* Wo     = (const float*)d_in[4];
    const float* g_attn = (const float*)d_in[5];
    const float* b_attn = (const float*)d_in[6];
    const float* W1     = (const float*)d_in[7];
    const float* b1     = (const float*)d_in[8];
    const float* W2     = (const float*)d_in[9];
    const float* b2     = (const float*)d_in[10];
    const float* g_ffn  = (const float*)d_in[11];
    const float* b_ffn  = (const float*)d_in[12];

    constexpr int M = 16384, D = 1024, FF = 4096;
    const size_t MD = (size_t)M * D;     // 16M elems
    const size_t DD = (size_t)D * D;     // 1M elems
    const size_t DF = (size_t)D * FF;    // 4M elems

    char* w = (char*)d_ws;
    bf16_t* Xb    = (bf16_t*)w; w += MD * 2;
    bf16_t* Wtq   = (bf16_t*)w; w += DD * 2;
    bf16_t* Wtk   = (bf16_t*)w; w += DD * 2;
    bf16_t* Wtv   = (bf16_t*)w; w += DD * 2;
    bf16_t* Wto   = (bf16_t*)w; w += DD * 2;
    bf16_t* Wt1   = (bf16_t*)w; w += DF * 2;   // [FF][D]
    bf16_t* Wt2   = (bf16_t*)w; w += DF * 2;   // [D][FF]
    bf16_t* qb    = (bf16_t*)w; w += MD * 2;
    bf16_t* kb    = (bf16_t*)w; w += MD * 2;
    bf16_t* vb    = (bf16_t*)w; w += MD * 2;
    bf16_t* attnb = (bf16_t*)w; w += MD * 2;
    bf16_t* hb    = qb;                        // reuse q/k/v/attn region (4*MD halves = M*FF)
    float*  projf = (float*)w;  w += MD * 4;
    float*  y2f   = projf;                     // reuse (proj dead after LN1)
    float*  x1f   = (float*)w;  w += MD * 4;
    bf16_t* x1b   = (bf16_t*)w; w += MD * 2;

    const dim3 blk(256);
    const dim3 gD(D / 128, M / 128);     // (8, 128)
    const dim3 gFF(FF / 128, M / 128);   // (32, 128)

    // One-time precision conversion / weight transposition
    convert_f32_bf16<<<(unsigned)(MD / 2048), blk, 0, stream>>>(X, Xb, (long long)MD);
    transpose_convert<<<dim3(16, 16), blk, 0, stream>>>(Wq, Wtq, D, D);
    transpose_convert<<<dim3(16, 16), blk, 0, stream>>>(Wk, Wtk, D, D);
    transpose_convert<<<dim3(16, 16), blk, 0, stream>>>(Wv, Wtv, D, D);
    transpose_convert<<<dim3(16, 16), blk, 0, stream>>>(Wo, Wto, D, D);
    transpose_convert<<<dim3(64, 16), blk, 0, stream>>>(W1, Wt1, D, FF);
    transpose_convert<<<dim3(16, 64), blk, 0, stream>>>(W2, Wt2, FF, D);

    // QKV projections (bf16 out); DH^-0.5 folded into Q alpha
    gemm_tdm_wmma<true><<<gD, blk, 0, stream>>>(Xb, Wtq, nullptr, qb, M, D, D, 0.125f, 0);
    gemm_tdm_wmma<true><<<gD, blk, 0, stream>>>(Xb, Wtk, nullptr, kb, M, D, D, 1.0f, 0);
    gemm_tdm_wmma<true><<<gD, blk, 0, stream>>>(Xb, Wtv, nullptr, vb, M, D, D, 1.0f, 0);
    // Local block attention
    local_attn_wmma<<<dim3(16, 64), blk, 0, stream>>>(qb, kb, vb, attnb);
    // Output projection (fp32 out for residual+LN)
    gemm_tdm_wmma<false><<<gD, blk, 0, stream>>>(attnb, Wto, nullptr, projf, M, D, D, 1.0f, 0);
    // Residual + LN 1 (fp32 + bf16 copies)
    add_layernorm<<<M, blk, 0, stream>>>(projf, X, g_attn, b_attn, x1f, x1b);
    // FFN
    gemm_tdm_wmma<true><<<gFF, blk, 0, stream>>>(x1b, Wt1, b1, hb, M, FF, D, 1.0f, 1);
    gemm_tdm_wmma<false><<<gD, blk, 0, stream>>>(hb, Wt2, b2, y2f, M, D, FF, 1.0f, 0);
    // Residual + LN 2 -> output (fp32)
    add_layernorm<<<M, blk, 0, stream>>>(y2f, x1f, g_ffn, b_ffn, (float*)d_out, nullptr);
}
